// RecurrentExecutionEngine_84765474554096
// MI455X (gfx1250) — matrix operand order repaired and checked by reading, not compiled, not run on hardware
//
#include <hip/hip_runtime.h>
#include <hip/hip_bf16.h>

typedef _Float16 half_t;
typedef __attribute__((ext_vector_type(8)))  _Float16 v8h;
typedef __attribute__((ext_vector_type(16))) _Float16 v16h;
typedef __attribute__((ext_vector_type(8)))  float    v8f;
typedef __attribute__((ext_vector_type(4)))  float    v4f;

#define NN 50000
#define EE 800000
#define BB 64
#define DD 128
#define DI_ 256
#define STEPS_ 5

// ---------------------------------------------------------------- WMMA helpers

__device__ __forceinline__ v8f wmma16(v16h a, v16h b, v8f c) {
  // D = A(16x32 f16) * B(32x16 f16) + C(16x16 f32)
  return __builtin_amdgcn_wmma_f32_16x16x32_f16(false, a, false, b, (short)0, c,
                                                false, false);
}

// A fragment from a row-major LDS tile (16 rows starting at `base`, leading dim
// ldm halfs). A uses contiguous K order: lane<16 -> M=lane, K=0..15;
// lane>=16 -> M=lane-16, K=16..31.
__device__ __forceinline__ v16h lds_a_frag(const half_t* base, int ldm, int lane,
                                           int k0) {
  int m  = lane & 15;
  int kb = (lane >> 4) << 4;              // 0 or 16
  const half_t* p = base + m * ldm + k0 + kb;
  v8h lo = *(const v8h*)p;                // ds_load_b128
  v8h hi = *(const v8h*)(p + 8);          // ds_load_b128
  v16h a;
#pragma unroll
  for (int i = 0; i < 8; ++i) { a[i] = lo[i]; a[i + 8] = hi[i]; }
  return a;
}

// B fragment from pre-packed weights: wf[nt][kt][lane][16] halfs, where lane<16
// holds N=lane,K=kt*32+0..15 and lane>=16 holds N=lane-16,K=kt*32+16..31
// (contiguous K order, the same one lds_a_frag uses).
__device__ __forceinline__ v16h b_frag(const half_t* w, int nkt, int nt, int kt,
                                       int lane) {
  const half_t* p = w + ((((size_t)nt * nkt + kt) * 32 + lane) << 4);
  v8h lo = *(const v8h*)p;                // global_load_b128
  v8h hi = *(const v8h*)(p + 8);          // global_load_b128
  v16h b;
#pragma unroll
  for (int i = 0; i < 8; ++i) { b[i] = lo[i]; b[i + 8] = hi[i]; }
  return b;
}

// ---------------------------------------------------------------- weight pack

__global__ void pack_w(const float* __restrict__ w, half_t* __restrict__ out,
                       int K, int Nw) {
  int idx = blockIdx.x * blockDim.x + threadIdx.x;
  if (idx >= K * Nw) return;
  int k = idx / Nw, n = idx - k * Nw;
  int kt = k >> 5, kr = k & 31;
  int nt = n >> 4, nr = n & 15;
  int lane = nr + ((kr & 16) ? 16 : 0);
  int elem = kr & 15;
  int nkt = K >> 5;
  out[((((size_t)nt * nkt + kt) * 32 + lane) << 4) + elem] = (half_t)w[idx];
}

// ---------------------------------------------------------------- misc kernels

__global__ void zero_kernel(float* p, long n) {
  long i = blockIdx.x * (long)blockDim.x + threadIdx.x;
  if (i < n) p[i] = 0.f;
}
__global__ void copy_kernel(float* dst, const float* src, long n) {
  long i = blockIdx.x * (long)blockDim.x + threadIdx.x;
  if (i < n) dst[i] = src[i];
}
__global__ void deg_kernel(const int* __restrict__ col, float* deg) {
  int e = blockIdx.x * blockDim.x + threadIdx.x;
  if (e < EE) atomicAdd(&deg[col[e]], 1.f);
}
__global__ void cnt_kernel(const int* __restrict__ batch, float* cnt) {
  int v = blockIdx.x * blockDim.x + threadIdx.x;
  if (v < NN) atomicAdd(&cnt[batch[v]], 1.f);
}

// ------------------------------------------------- edge message MLP + scatter
// msg = relu([x[row]|hist[batch[row]]] @ W1 + b1) @ W2 + b2 ; agg[col] += msg

__global__ __launch_bounds__(128) void edge_msg_kernel(
    const float* __restrict__ x, const float* __restrict__ hist,
    const int* __restrict__ row, const int* __restrict__ col,
    const int* __restrict__ batch,
    const half_t* __restrict__ w1, const float* __restrict__ b1,
    const half_t* __restrict__ w2, const float* __restrict__ b2,
    float* __restrict__ agg) {
  __shared__ __align__(16) half_t in_t[64][264];
  __shared__ __align__(16) half_t hid_t[64][136];
  const int tid = threadIdx.x, lane = tid & 31, wave = tid >> 5;
  const int e0 = blockIdx.x * 64;
  {  // gather: 2 threads per edge, 128 cols each
    int le = tid >> 1, hs = tid & 1, e = e0 + le;
    half_t* dst = &in_t[le][hs * 128];
    if (e < EE) {
      int r = row[e];
      const float* src = hs ? (hist + (size_t)batch[r] * DD)
                            : (x + (size_t)r * DD);
      const v4f* s4 = (const v4f*)src;
#pragma unroll
      for (int i = 0; i < 32; ++i) {
        v4f v = s4[i];
        dst[4 * i + 0] = (half_t)v[0]; dst[4 * i + 1] = (half_t)v[1];
        dst[4 * i + 2] = (half_t)v[2]; dst[4 * i + 3] = (half_t)v[3];
      }
    } else {
      for (int i = 0; i < 128; ++i) dst[i] = (half_t)0.f;
    }
  }
  __syncthreads();
  const int m0 = wave * 16;
  {  // layer 1: K=256
    v16h a[8];
#pragma unroll
    for (int k = 0; k < 8; ++k) a[k] = lds_a_frag(&in_t[m0][0], 264, lane, k * 32);
    for (int nt = 0; nt < 8; ++nt) {
      v8f c = {};
#pragma unroll
      for (int k = 0; k < 8; ++k) c = wmma16(a[k], b_frag(w1, 8, nt, k, lane), c);
      int n = nt * 16 + (lane & 15);
      float bias = b1[n];
      int mb = m0 + ((lane >> 4) << 3);
#pragma unroll
      for (int r = 0; r < 8; ++r) {
        float v = c[r] + bias;
        hid_t[mb + r][n] = (half_t)(v > 0.f ? v : 0.f);
      }
    }
  }
  __syncthreads();
  {  // layer 2: K=128 -> scatter with atomics
    v16h a[4];
#pragma unroll
    for (int k = 0; k < 4; ++k) a[k] = lds_a_frag(&hid_t[m0][0], 136, lane, k * 32);
    for (int nt = 0; nt < 8; ++nt) {
      v8f c = {};
#pragma unroll
      for (int k = 0; k < 4; ++k) c = wmma16(a[k], b_frag(w2, 4, nt, k, lane), c);
      int n = nt * 16 + (lane & 15);
      float bias = b2[n];
      int mb = m0 + ((lane >> 4) << 3);
#pragma unroll
      for (int r = 0; r < 8; ++r) {
        int e = e0 + mb + r;
        if (e < EE) atomicAdd(&agg[(size_t)col[e] * DD + n], c[r] + bias);
      }
    }
  }
}

// ------------------------------------------------------------ node update MLP
// x_out = relu([x|agg/deg|u[batch]] @ W1 + b1) @ W2 + b2 + x  (in-place nodebuf)
// K=512 processed in two staged halves to keep LDS <= 51KB.

__global__ __launch_bounds__(128) void node_update_kernel(
    const float* __restrict__ x, const float* __restrict__ u,
    const int* __restrict__ batch,
    const half_t* __restrict__ w1, const float* __restrict__ b1,
    const half_t* __restrict__ w2, const float* __restrict__ b2,
    const float* __restrict__ deg, float* __restrict__ nodebuf) {
  __shared__ __align__(16) half_t in_t[64][264];
  __shared__ __align__(16) half_t hid_t[64][136];
  const int tid = threadIdx.x, lane = tid & 31, wave = tid >> 5;
  const int v0 = blockIdx.x * 64;
  const int m0 = wave * 16;
  {  // phase A stage: cols 0..255 = [x | agg/deg]
    int lv = tid >> 1, hs = tid & 1, v = v0 + lv;
    half_t* dst = &in_t[lv][hs * 128];
    if (v < NN) {
      if (hs == 0) {
        const v4f* s4 = (const v4f*)(x + (size_t)v * DD);
#pragma unroll
        for (int i = 0; i < 32; ++i) {
          v4f t = s4[i];
          dst[4 * i + 0] = (half_t)t[0]; dst[4 * i + 1] = (half_t)t[1];
          dst[4 * i + 2] = (half_t)t[2]; dst[4 * i + 3] = (half_t)t[3];
        }
      } else {
        float inv = 1.f / fmaxf(deg[v], 1.f);
        const v4f* s4 = (const v4f*)(nodebuf + (size_t)v * DD);
#pragma unroll
        for (int i = 0; i < 32; ++i) {
          v4f t = s4[i];
          dst[4 * i + 0] = (half_t)(t[0] * inv); dst[4 * i + 1] = (half_t)(t[1] * inv);
          dst[4 * i + 2] = (half_t)(t[2] * inv); dst[4 * i + 3] = (half_t)(t[3] * inv);
        }
      }
    } else {
      for (int i = 0; i < 128; ++i) dst[i] = (half_t)0.f;
    }
  }
  __syncthreads();
  v8f acc[8] = {};
  {  // layer1 K-steps 0..7
    v16h a[8];
#pragma unroll
    for (int k = 0; k < 8; ++k) a[k] = lds_a_frag(&in_t[m0][0], 264, lane, k * 32);
    for (int nt = 0; nt < 8; ++nt)
#pragma unroll
      for (int k = 0; k < 8; ++k)
        acc[nt] = wmma16(a[k], b_frag(w1, 16, nt, k, lane), acc[nt]);
  }
  __syncthreads();
  {  // phase B stage: cols 256..511 = u[batch[v]]
    int lv = tid >> 1, hs = tid & 1, v = v0 + lv;
    half_t* dst = &in_t[lv][hs * 128];
    if (v < NN) {
      const v4f* s4 = (const v4f*)(u + (size_t)batch[v] * DI_ + hs * 128);
#pragma unroll
      for (int i = 0; i < 32; ++i) {
        v4f t = s4[i];
        dst[4 * i + 0] = (half_t)t[0]; dst[4 * i + 1] = (half_t)t[1];
        dst[4 * i + 2] = (half_t)t[2]; dst[4 * i + 3] = (half_t)t[3];
      }
    } else {
      for (int i = 0; i < 128; ++i) dst[i] = (half_t)0.f;
    }
  }
  __syncthreads();
  {  // layer1 K-steps 8..15, then relu+bias into hid
    v16h a[8];
#pragma unroll
    for (int k = 0; k < 8; ++k) a[k] = lds_a_frag(&in_t[m0][0], 264, lane, k * 32);
    for (int nt = 0; nt < 8; ++nt) {
#pragma unroll
      for (int k = 0; k < 8; ++k)
        acc[nt] = wmma16(a[k], b_frag(w1, 16, nt, 8 + k, lane), acc[nt]);
      int n = nt * 16 + (lane & 15);
      float bias = b1[n];
      int mb = m0 + ((lane >> 4) << 3);
#pragma unroll
      for (int r = 0; r < 8; ++r) {
        float v = acc[nt][r] + bias;
        hid_t[mb + r][n] = (half_t)(v > 0.f ? v : 0.f);
      }
    }
  }
  __syncthreads();
  {  // layer2 K=128, epilogue adds residual x and writes nodebuf in place
    v16h a[4];
#pragma unroll
    for (int k = 0; k < 4; ++k) a[k] = lds_a_frag(&hid_t[m0][0], 136, lane, k * 32);
    for (int nt = 0; nt < 8; ++nt) {
      v8f c = {};
#pragma unroll
      for (int k = 0; k < 4; ++k) c = wmma16(a[k], b_frag(w2, 4, nt, k, lane), c);
      int n = nt * 16 + (lane & 15);
      float bias = b2[n];
      int mb = m0 + ((lane >> 4) << 3);
#pragma unroll
      for (int r = 0; r < 8; ++r) {
        int v = v0 + mb + r;
        if (v < NN)
          nodebuf[(size_t)v * DD + n] = c[r] + bias + x[(size_t)v * DD + n];
      }
    }
  }
}

// ------------------------------------------------------------- graph layernorm

__global__ void ln_sum_kernel(const float* __restrict__ nb,
                              const int* __restrict__ batch, float* gsum) {
  int v = blockIdx.x * blockDim.x + threadIdx.x;
  if (v >= NN) return;
  const v4f* p = (const v4f*)(nb + (size_t)v * DD);
  float s = 0.f;
#pragma unroll
  for (int i = 0; i < 32; ++i) { v4f t = p[i]; s += t[0] + t[1] + t[2] + t[3]; }
  atomicAdd(&gsum[batch[v]], s);
}
__global__ void ln_var_kernel(const float* __restrict__ nb,
                              const int* __restrict__ batch,
                              const float* __restrict__ gsum,
                              const float* __restrict__ cnt, float* gvar) {
  int v = blockIdx.x * blockDim.x + threadIdx.x;
  if (v >= NN) return;
  int g = batch[v];
  float mean = gsum[g] / (fmaxf(cnt[g], 1.f) * 128.f);
  const v4f* p = (const v4f*)(nb + (size_t)v * DD);
  float s = 0.f;
#pragma unroll
  for (int i = 0; i < 32; ++i) {
    v4f t = p[i];
#pragma unroll
    for (int j = 0; j < 4; ++j) { float d = t[j] - mean; s += d * d; }
  }
  atomicAdd(&gvar[g], s);
}
__global__ void ln_apply_kernel(float* nb, const int* __restrict__ batch,
                                const float* __restrict__ gsum,
                                const float* __restrict__ gvar,
                                const float* __restrict__ cnt,
                                const float* __restrict__ lnw,
                                const float* __restrict__ lnb) {
  long idx = blockIdx.x * (long)blockDim.x + threadIdx.x;
  if (idx >= (long)NN * DD) return;
  int v = (int)(idx >> 7), d = (int)(idx & 127);
  int g = batch[v];
  float c = fmaxf(cnt[g], 1.f) * 128.f;
  float mean = gsum[g] / c;
  float var = gvar[g] / c;
  nb[idx] = (nb[idx] - mean) * rsqrtf(var + 1e-5f) * lnw[d] + lnb[d];
}

// ------------------------------------------------------------------- gate MLP
// gate = relu(x_ln @ Wg1 + bg1) @ wg2 + bg2   (layer2 is a 128-dot per node)

__global__ __launch_bounds__(128) void gate_kernel(
    const float* __restrict__ nodebuf, const half_t* __restrict__ wg1,
    const float* __restrict__ bg1, const float* __restrict__ wg2,
    const float* __restrict__ bg2, float* __restrict__ gate) {
  __shared__ __align__(16) half_t in_t[64][136];
  __shared__ __align__(16) half_t hid_t[64][136];
  const int tid = threadIdx.x, lane = tid & 31, wave = tid >> 5;
  const int v0 = blockIdx.x * 64;
  {
    int lv = tid >> 1, hs = tid & 1, v = v0 + lv;
    half_t* dst = &in_t[lv][hs * 64];
    if (v < NN) {
      const v4f* s4 = (const v4f*)(nodebuf + (size_t)v * DD + hs * 64);
#pragma unroll
      for (int i = 0; i < 16; ++i) {
        v4f t = s4[i];
        dst[4 * i + 0] = (half_t)t[0]; dst[4 * i + 1] = (half_t)t[1];
        dst[4 * i + 2] = (half_t)t[2]; dst[4 * i + 3] = (half_t)t[3];
      }
    } else {
      for (int i = 0; i < 64; ++i) dst[i] = (half_t)0.f;
    }
  }
  __syncthreads();
  const int m0 = wave * 16;
  {
    v16h a[4];
#pragma unroll
    for (int k = 0; k < 4; ++k) a[k] = lds_a_frag(&in_t[m0][0], 136, lane, k * 32);
    for (int nt = 0; nt < 8; ++nt) {
      v8f c = {};
#pragma unroll
      for (int k = 0; k < 4; ++k) c = wmma16(a[k], b_frag(wg1, 4, nt, k, lane), c);
      int n = nt * 16 + (lane & 15);
      float bias = bg1[n];
      int mb = m0 + ((lane >> 4) << 3);
#pragma unroll
      for (int r = 0; r < 8; ++r) {
        float v = c[r] + bias;
        hid_t[mb + r][n] = (half_t)(v > 0.f ? v : 0.f);
      }
    }
  }
  __syncthreads();
  if (tid < 64) {
    int v = v0 + tid;
    if (v < NN) {
      float s = bg2[0];
      for (int d = 0; d < DD; ++d) s += (float)hid_t[tid][d] * wg2[d];
      gate[v] = s;
    }
  }
}

// ---------------------------------------------- per-graph softmax + hist accum

__global__ __launch_bounds__(256) void softmax_hist_kernel(
    const float* __restrict__ nodebuf, float* __restrict__ gate,
    const int* __restrict__ batch, float* __restrict__ hist,
    float* __restrict__ hists_step, float* __restrict__ bitmap, int step) {
  __shared__ float red[256];
  const int b = blockIdx.x, tid = threadIdx.x;
  int lo = 0, hi = NN;
  while (lo < hi) { int mid = (lo + hi) >> 1; if (batch[mid] < b) lo = mid + 1; else hi = mid; }
  const int start = lo;
  hi = NN;
  while (lo < hi) { int mid = (lo + hi) >> 1; if (batch[mid] < b + 1) lo = mid + 1; else hi = mid; }
  const int end = lo;
  // max
  float m = -3.4e38f;
  for (int v = start + tid; v < end; v += 256) m = fmaxf(m, gate[v]);
  red[tid] = m; __syncthreads();
  for (int s = 128; s > 0; s >>= 1) { if (tid < s) red[tid] = fmaxf(red[tid], red[tid + s]); __syncthreads(); }
  m = red[0]; __syncthreads();
  // sum of exp
  float s = 0.f;
  for (int v = start + tid; v < end; v += 256) s += expf(gate[v] - m);
  red[tid] = s; __syncthreads();
  for (int t = 128; t > 0; t >>= 1) { if (tid < t) red[tid] += red[tid + t]; __syncthreads(); }
  const float denom = red[0] + 1e-16f;
  __syncthreads();
  // normalize + bitmap
  for (int v = start + tid; v < end; v += 256) {
    float g = expf(gate[v] - m) / denom;
    gate[v] = g;
    bitmap[(size_t)v * STEPS_ + step] = g;
  }
  __syncthreads();
  // hist[b] = sum_v gate[v] * x_out[v]
  if (tid < DD) {
    float acc = 0.f;
    for (int v = start; v < end; ++v)
      acc += gate[v] * nodebuf[(size_t)v * DD + tid];
    hist[(size_t)b * DD + tid] = acc;
    hists_step[(size_t)b * DD + tid] = acc;
  }
}

// ------------------------------------------------------------ history MLP (end)
// [STEPS*B,128] -> relu(@Wh1 128x256) -> @Wh2 256x256 -> out [STEPS,B,256]

__global__ __launch_bounds__(128) void hist_mlp_kernel(
    const float* __restrict__ hists, const half_t* __restrict__ w1,
    const float* __restrict__ b1, const half_t* __restrict__ w2,
    const float* __restrict__ b2, float* __restrict__ out) {
  __shared__ __align__(16) half_t in_t[64][136];
  __shared__ __align__(16) half_t hid_t[64][264];
  const int R = STEPS_ * BB;  // 320
  const int tid = threadIdx.x, lane = tid & 31, wave = tid >> 5;
  const int r0 = blockIdx.x * 64;
  {
    int lr = tid >> 1, hs = tid & 1, r = r0 + lr;
    half_t* dst = &in_t[lr][hs * 64];
    if (r < R) {
      const v4f* s4 = (const v4f*)(hists + (size_t)r * DD + hs * 64);
#pragma unroll
      for (int i = 0; i < 16; ++i) {
        v4f t = s4[i];
        dst[4 * i + 0] = (half_t)t[0]; dst[4 * i + 1] = (half_t)t[1];
        dst[4 * i + 2] = (half_t)t[2]; dst[4 * i + 3] = (half_t)t[3];
      }
    } else {
      for (int i = 0; i < 64; ++i) dst[i] = (half_t)0.f;
    }
  }
  __syncthreads();
  const int m0 = wave * 16;
  {
    v16h a[4];
#pragma unroll
    for (int k = 0; k < 4; ++k) a[k] = lds_a_frag(&in_t[m0][0], 136, lane, k * 32);
    for (int nt = 0; nt < 16; ++nt) {
      v8f c = {};
#pragma unroll
      for (int k = 0; k < 4; ++k) c = wmma16(a[k], b_frag(w1, 4, nt, k, lane), c);
      int n = nt * 16 + (lane & 15);
      float bias = b1[n];
      int mb = m0 + ((lane >> 4) << 3);
#pragma unroll
      for (int r = 0; r < 8; ++r) {
        float v = c[r] + bias;
        hid_t[mb + r][n] = (half_t)(v > 0.f ? v : 0.f);
      }
    }
  }
  __syncthreads();
  {
    v16h a[8];
#pragma unroll
    for (int k = 0; k < 8; ++k) a[k] = lds_a_frag(&hid_t[m0][0], 264, lane, k * 32);
    for (int nt = 0; nt < 16; ++nt) {
      v8f c = {};
#pragma unroll
      for (int k = 0; k < 8; ++k) c = wmma16(a[k], b_frag(w2, 8, nt, k, lane), c);
      int n = nt * 16 + (lane & 15);
      float bias = b2[n];
      int mb = m0 + ((lane >> 4) << 3);
#pragma unroll
      for (int rr = 0; rr < 8; ++rr) {
        int r = r0 + mb + rr;
        if (r < R) out[(size_t)r * DI_ + n] = c[rr] + bias;
      }
    }
  }
}

// ----------------------------------------------------------------------- host

extern "C" void kernel_launch(void* const* d_in, const int* in_sizes, int n_in,
                              void* d_out, int out_size, void* d_ws,
                              size_t ws_size, hipStream_t stream) {
  (void)in_sizes; (void)n_in; (void)out_size; (void)ws_size;
  const float* x     = (const float*)d_in[0];
  const float* instr = (const float*)d_in[2];   // d_in[1] edge_attr unused
  const float* w_n1a = (const float*)d_in[3];  const float* b_n1a = (const float*)d_in[4];
  const float* w_n1b = (const float*)d_in[5];  const float* b_n1b = (const float*)d_in[6];
  const float* w_n2a = (const float*)d_in[7];  const float* b_n2a = (const float*)d_in[8];
  const float* w_n2b = (const float*)d_in[9];  const float* b_n2b = (const float*)d_in[10];
  const float* ln_w  = (const float*)d_in[11]; const float* ln_b  = (const float*)d_in[12];
  const float* w_g1  = (const float*)d_in[13]; const float* b_g1  = (const float*)d_in[14];
  const float* w_g2  = (const float*)d_in[15]; const float* b_g2  = (const float*)d_in[16];
  const float* w_h1  = (const float*)d_in[17]; const float* b_h1  = (const float*)d_in[18];
  const float* w_h2  = (const float*)d_in[19]; const float* b_h2  = (const float*)d_in[20];
  const int* edge_index = (const int*)d_in[21];
  const int* batch      = (const int*)d_in[22];
  const int* row = edge_index;
  const int* col = edge_index + EE;

  float* out        = (float*)d_out;
  float* out_bitmap = out + (size_t)NN * DD;
  float* out_hist   = out_bitmap + (size_t)NN * STEPS_;

  char* ws = (char*)d_ws;
  size_t off = 0;
  auto alloc = [&](size_t bytes) -> char* {
    char* p = ws + off;
    off = (off + bytes + 255) & ~(size_t)255;
    return p;
  };
  half_t* pw1a = (half_t*)alloc(256 * 128 * 2);
  half_t* pw1b = (half_t*)alloc(128 * 128 * 2);
  half_t* pw2a = (half_t*)alloc(512 * 128 * 2);
  half_t* pw2b = (half_t*)alloc(128 * 128 * 2);
  half_t* pwg1 = (half_t*)alloc(128 * 128 * 2);
  half_t* pwh1 = (half_t*)alloc(128 * 256 * 2);
  half_t* pwh2 = (half_t*)alloc(256 * 256 * 2);
  float* deg   = (float*)alloc((size_t)NN * 4);
  float* cnt   = (float*)alloc(BB * 4);
  float* gsum  = (float*)alloc(BB * 4);
  float* gvar  = (float*)alloc(BB * 4);
  float* hist  = (float*)alloc((size_t)BB * DD * 4);
  float* hists = (float*)alloc((size_t)STEPS_ * BB * DD * 4);
  float* gate  = (float*)alloc((size_t)NN * 4);
  float* nodebuf = (float*)alloc((size_t)NN * DD * 4);

  auto cdiv = [](long a, long b) -> unsigned { return (unsigned)((a + b - 1) / b); };

  // pack weights f32 -> f16 B-fragment layout
  pack_w<<<cdiv(256 * 128, 256), 256, 0, stream>>>(w_n1a, pw1a, 256, 128);
  pack_w<<<cdiv(128 * 128, 256), 256, 0, stream>>>(w_n1b, pw1b, 128, 128);
  pack_w<<<cdiv(512 * 128, 256), 256, 0, stream>>>(w_n2a, pw2a, 512, 128);
  pack_w<<<cdiv(128 * 128, 256), 256, 0, stream>>>(w_n2b, pw2b, 128, 128);
  pack_w<<<cdiv(128 * 128, 256), 256, 0, stream>>>(w_g1, pwg1, 128, 128);
  pack_w<<<cdiv(128 * 256, 256), 256, 0, stream>>>(w_h1, pwh1, 128, 256);
  pack_w<<<cdiv(256 * 256, 256), 256, 0, stream>>>(w_h2, pwh2, 256, 256);

  // init
  zero_kernel<<<cdiv(NN, 256), 256, 0, stream>>>(deg, NN);
  zero_kernel<<<1, 256, 0, stream>>>(cnt, BB);
  zero_kernel<<<cdiv((long)BB * DD, 256), 256, 0, stream>>>(hist, (long)BB * DD);
  deg_kernel<<<cdiv(EE, 256), 256, 0, stream>>>(col, deg);
  cnt_kernel<<<cdiv(NN, 256), 256, 0, stream>>>(batch, cnt);
  copy_kernel<<<cdiv((long)NN * DD, 256), 256, 0, stream>>>(out, x, (long)NN * DD);

  for (int step = 0; step < STEPS_; ++step) {
    zero_kernel<<<cdiv((long)NN * DD, 256), 256, 0, stream>>>(nodebuf, (long)NN * DD);
    zero_kernel<<<1, 256, 0, stream>>>(gsum, BB);
    zero_kernel<<<1, 256, 0, stream>>>(gvar, BB);
    edge_msg_kernel<<<cdiv(EE, 64), 128, 0, stream>>>(
        x, hist, row, col, batch, pw1a, b_n1a, pw1b, b_n1b, nodebuf);
    node_update_kernel<<<cdiv(NN, 64), 128, 0, stream>>>(
        x, instr + (size_t)step * BB * DI_, batch, pw2a, b_n2a, pw2b, b_n2b,
        deg, nodebuf);
    ln_sum_kernel<<<cdiv(NN, 256), 256, 0, stream>>>(nodebuf, batch, gsum);
    ln_var_kernel<<<cdiv(NN, 256), 256, 0, stream>>>(nodebuf, batch, gsum, cnt, gvar);
    ln_apply_kernel<<<cdiv((long)NN * DD, 256), 256, 0, stream>>>(
        nodebuf, batch, gsum, gvar, cnt, ln_w, ln_b);
    gate_kernel<<<cdiv(NN, 64), 128, 0, stream>>>(nodebuf, pwg1, b_g1, w_g2, b_g2, gate);
    softmax_hist_kernel<<<BB, 256, 0, stream>>>(
        nodebuf, gate, batch, hist, hists + (size_t)step * BB * DD, out_bitmap, step);
  }
  hist_mlp_kernel<<<cdiv(STEPS_ * BB, 64), 128, 0, stream>>>(
      hists, pwh1, b_h1, pwh2, b_h2, out_hist);
}
